// DeviceAttention_60705067762179
// MI455X (gfx1250) — matrix operand order repaired and checked by reading, not compiled, not run on hardware
//
#include <hip/hip_runtime.h>

typedef __attribute__((ext_vector_type(16))) __bf16 v16bf;
typedef __attribute__((ext_vector_type(8)))  __bf16 v8bf;
typedef __attribute__((ext_vector_type(8)))  float  v8f;

#define TT  2048
#define CC  512
#define MT  32          // tokens per workgroup
#define XST 72          // padded LDS row stride (bf16 elems) for 64-col tiles
#define CST 40          // padded LDS row stride for 32-col cat tile

// ---- LDS byte offsets (dynamic shared memory partition) ----
#define OFF_XE   0                          // 512*72*2 = 73728   (later reused as attended^T)
#define OFF_QKV  73728                      // 384*72*2 = 55296
#define OFF_CTX  129024                     // 512*72*2 = 73728
#define OFF_CAT  73728                      // 1024*40*2 = 81920 (aliases dead qkv+ctx)
#define OFF_BIAS 202752                     // 3584 floats = 14336
#define OFF_PRB  217088                     // 128 floats = 512
#define OFF_AVG  217600                     // 4 floats
#define SMEM_BYTES 217616

// A-fragment: 16x32 bf16, row-major weight W[row][k], stride in elements.
__device__ __forceinline__ v16bf ldA(const __bf16* __restrict__ W, int stride,
                                     int row0, int kb, int lane) {
    const __bf16* p = W + (size_t)(row0 + (lane & 15)) * stride + kb + ((lane >> 4) << 3);
    v8bf lo = *(const v8bf*)p;          // K = kb + 8*half .. +8
    v8bf hi = *(const v8bf*)(p + 16);   // K = kb + 16 + 8*half .. +8
    return __builtin_shufflevector(lo, hi, 0,1,2,3,4,5,6,7,8,9,10,11,12,13,14,15);
}

// B-fragment: 32x16 bf16 from LDS laid out [K][N] with given stride; lane holds column N=n0+lane%16, K=kb+8*half+s / kb+16+8*half+(s-8).
__device__ __forceinline__ v16bf ldB(const __bf16* __restrict__ Bm, int stride,
                                     int kb, int n0, int lane) {
    const __bf16* p = Bm + (size_t)(kb + ((lane >> 4) << 3)) * stride + n0 + (lane & 15);
    v16bf r;
#pragma unroll
    for (int s = 0; s < 8; ++s) {
        r[s]     = p[s * stride];
        r[8 + s] = p[(16 + s) * stride];
    }
    return r;
}

__device__ __forceinline__ v8f wmma_bf16(v16bf a, v16bf b, v8f c) {
    return __builtin_amdgcn_wmma_f32_16x16x32_bf16(false, a, false, b, (short)0, c, false, false);
}

__global__ void cvt_bf16_kernel(const float* __restrict__ src, __bf16* __restrict__ dst, int n) {
    int i = (blockIdx.x * 256 + threadIdx.x) * 4;
    if (i < n) {
        float4 v = *(const float4*)(src + i);
        dst[i + 0] = (__bf16)v.x;
        dst[i + 1] = (__bf16)v.y;
        dst[i + 2] = (__bf16)v.z;
        dst[i + 3] = (__bf16)v.w;
    }
}

__global__ void zero_avg_kernel(float* __restrict__ avg) {
    if (threadIdx.x < 64) avg[threadIdx.x] = 0.0f;
}

__global__ __launch_bounds__(256)
void fused_devattn_kernel(const float* __restrict__ x,
                          const float* __restrict__ demb,
                          const __bf16* __restrict__ Wqkv,   // [1536][512]
                          const __bf16* __restrict__ Wout,   // [512][512]
                          const __bf16* __restrict__ Wph,    // [512][512]
                          const __bf16* __restrict__ Wim,    // [512][512]
                          const __bf16* __restrict__ Wo2,    // [512][1024]
                          const float* __restrict__ b_qkv,
                          const float* __restrict__ b_out,
                          const float* __restrict__ b_ph,
                          const float* __restrict__ b_im,
                          const float* __restrict__ b_o2,
                          float* __restrict__ out,
                          float* __restrict__ avg) {
    extern __shared__ char smem[];
    __bf16* xeT   = (__bf16*)(smem + OFF_XE);    // [512][XST] cols: d*32+t  (later attended^T)
    __bf16* qkvT  = (__bf16*)(smem + OFF_QKV);   // [384][XST] q:0-127 k:128-255 v:256-383
    __bf16* ctxT  = (__bf16*)(smem + OFF_CTX);   // [512][XST]
    __bf16* catT  = (__bf16*)(smem + OFF_CAT);   // [1024][CST] phone rows 0-511, imu 512-1023
    float*  biasL = (float*)(smem + OFF_BIAS);   // qkv[0,1536) out[1536,2048) ph[2048,2560) im[2560,3072) o2[3072,3584)
    float*  prbL  = (float*)(smem + OFF_PRB);    // probs[t][dq][dk] fp32 (current head)
    float*  avgL  = (float*)(smem + OFF_AVG);    // 2x2 partial sums

    const int tid  = threadIdx.x;
    const int lane = tid & 31;
    const int w    = tid >> 5;
    const int hh   = lane >> 4;       // half-select
    const int l15  = lane & 15;
    const int bidx = blockIdx.x;
    const int b    = bidx >> 6;                  // 64 t-tiles per batch
    const int t0   = (bidx & 63) * MT;

    // ---- preload biases to LDS ----
    for (int i = tid; i < 3584; i += 256) {
        float v;
        if      (i < 1536) v = b_qkv[i];
        else if (i < 2048) v = b_out[i - 1536];
        else if (i < 2560) v = b_ph[i - 2048];
        else if (i < 3072) v = b_im[i - 2560];
        else               v = b_o2[i - 3072];
        biasL[i] = v;
    }
    if (tid < 4) avgL[tid] = 0.0f;

    // ---- stage 0: xe^T = x + device_embedding, bf16, LDS [c][d*32+t] ----
    const size_t xbase = ((size_t)b * 1024) * (size_t)TT + (size_t)t0;
    for (int j = tid; j < 8192; j += 256) {
        int r  = j >> 3;               // 0..1023 == c*2 + d
        int q4 = (j & 7) << 2;         // 0..28
        float4 xv = *(const float4*)(x + xbase + (size_t)r * TT + q4);
        int c = r >> 1, d = r & 1;
        float ev = demb[d * CC + c];
        __bf16* dst = xeT + c * XST + d * MT + q4;
        dst[0] = (__bf16)(xv.x + ev);
        dst[1] = (__bf16)(xv.y + ev);
        dst[2] = (__bf16)(xv.z + ev);
        dst[3] = (__bf16)(xv.w + ev);
    }
    __syncthreads();

    float a0 = 0.0f, a1 = 0.0f;                 // avg-attention partials (threads < 64)
    const float scale = 0.08838834764831845f;   // 1/sqrt(128)

    for (int h = 0; h < 4; ++h) {
        // ---- stage 1: q/k/v^T for head h (WMMA). wave w owns row-strips r=w of q,k,v ----
        for (int i = 0; i < 3; ++i) {
            int wrow0 = i * CC + h * 128 + (w << 4);   // in_proj row base
            int lrow0 = i * 128 + (w << 4);            // qkvT row base
            v8f acc[4] = {};
            for (int kb = 0; kb < CC; kb += 32) {
                v16bf a = ldA(Wqkv, CC, wrow0, kb, lane);
#pragma unroll
                for (int nt = 0; nt < 4; ++nt) {
                    v16bf bm = ldB(xeT, XST, kb, nt * 16, lane);
                    acc[nt] = wmma_bf16(a, bm, acc[nt]);
                }
            }
#pragma unroll
            for (int g = 0; g < 8; ++g) {
                int ro = g + (hh << 3);
                float bv = biasL[wrow0 + ro];
                int lr = lrow0 + ro;
#pragma unroll
                for (int nt = 0; nt < 4; ++nt)
                    qkvT[lr * XST + nt * 16 + l15] = (__bf16)(acc[nt][g] + bv);
            }
        }
        __syncthreads();

        // ---- stage 2a: 2x2 scores + softmax (VALU) ----
        if (tid < 64) {
            int t = tid & 31, dq = tid >> 5;
            float s0 = 0.0f, s1 = 0.0f;
            for (int j = 0; j < 128; ++j) {
                float qv = (float)qkvT[j * XST + dq * MT + t];
                s0 += qv * (float)qkvT[(128 + j) * XST + t];
                s1 += qv * (float)qkvT[(128 + j) * XST + MT + t];
            }
            s0 *= scale; s1 *= scale;
            float m  = fmaxf(s0, s1);
            float e0 = __expf(s0 - m), e1 = __expf(s1 - m);
            float inv = 1.0f / (e0 + e1);
            float p0 = e0 * inv, p1 = e1 * inv;
            prbL[t * 4 + dq * 2 + 0] = p0;
            prbL[t * 4 + dq * 2 + 1] = p1;
            a0 += p0; a1 += p1;
        }
        __syncthreads();

        // ---- stage 2b: ctx^T[h*128+j][dq*32+t] = p0*v0 + p1*v1 ----
        {
            int c = tid & 63;
            int jr = tid >> 6;                   // 0..3
            int t = c & 31, dq = c >> 5;
            float p0 = prbL[t * 4 + dq * 2 + 0];
            float p1 = prbL[t * 4 + dq * 2 + 1];
            for (int jj = 0; jj < 32; ++jj) {
                int j = jr + (jj << 2);
                float v0 = (float)qkvT[(256 + j) * XST + t];
                float v1 = (float)qkvT[(256 + j) * XST + MT + t];
                ctxT[(h * 128 + j) * XST + c] = (__bf16)(p0 * v0 + p1 * v1);
            }
        }
        __syncthreads();
    }

    // reduce avg-attention partials into LDS (ds_add_f32)
    if (tid < 64) {
        int dq = tid >> 5;
        atomicAdd(&avgL[dq * 2 + 0], a0);
        atomicAdd(&avgL[dq * 2 + 1], a1);
    }

    // ---- stage 3: attended^T = Wout @ ctx^T + b  (written into xeT region) ----
    __bf16* attT = xeT;
    for (int i = 0; i < 4; ++i) {
        int o0 = ((w << 2) + i) << 4;
        v8f acc[4] = {};
        for (int kb = 0; kb < CC; kb += 32) {
            v16bf a = ldA(Wout, CC, o0, kb, lane);
#pragma unroll
            for (int nt = 0; nt < 4; ++nt) {
                v16bf bm = ldB(ctxT, XST, kb, nt * 16, lane);
                acc[nt] = wmma_bf16(a, bm, acc[nt]);
            }
        }
#pragma unroll
        for (int g = 0; g < 8; ++g) {
            int ro = g + (hh << 3);
            float bv = biasL[1536 + o0 + ro];
#pragma unroll
            for (int nt = 0; nt < 4; ++nt)
                attT[(o0 + ro) * XST + nt * 16 + l15] = (__bf16)(acc[nt][g] + bv);
        }
    }
    __syncthreads();

    // flush avg-attention to global (pre-zeroed); mean over H*T
    if (tid < 4) atomicAdd(&avg[b * 4 + tid], avgL[tid] * (1.0f / 8192.0f));

    // ---- stage 4: phone/imu -> cat^T [1024][32] ----
    for (int i = 0; i < 8; ++i) {
        int s = (w << 3) + i;                    // 0..63
        const __bf16* W = (s < 32) ? Wph : Wim;
        int o0 = (s & 31) << 4;
        int d  = s >> 5;
        int boff = 2048 + (d << 9) + o0;
        v8f acc[2] = {};
        for (int kb = 0; kb < CC; kb += 32) {
            v16bf a = ldA(W, CC, o0, kb, lane);
#pragma unroll
            for (int nt = 0; nt < 2; ++nt) {
                v16bf bm = ldB(attT, XST, kb, d * MT + nt * 16, lane);
                acc[nt] = wmma_bf16(a, bm, acc[nt]);
            }
        }
#pragma unroll
        for (int g = 0; g < 8; ++g) {
            int ro = g + (hh << 3);
            float bv = biasL[boff + ro];
#pragma unroll
            for (int nt = 0; nt < 2; ++nt)
                catT[(s * 16 + ro) * CST + nt * 16 + l15] = (__bf16)(acc[nt][g] + bv);
        }
    }
    __syncthreads();

    // ---- stage 5: out^T = Wo2 @ cat^T + b -> global (B,C,T) ----
    const size_t obase = (size_t)b * CC * TT + (size_t)t0;
    for (int i = 0; i < 4; ++i) {
        int o0 = ((w << 2) + i) << 4;
        v8f acc[2] = {};
        for (int kb = 0; kb < 1024; kb += 32) {
            v16bf a = ldA(Wo2, 1024, o0, kb, lane);
#pragma unroll
            for (int nt = 0; nt < 2; ++nt) {
                v16bf bm = ldB(catT, CST, kb, nt * 16, lane);
                acc[nt] = wmma_bf16(a, bm, acc[nt]);
            }
        }
#pragma unroll
        for (int g = 0; g < 8; ++g) {
            int ro = g + (hh << 3);
            float bv = biasL[3072 + o0 + ro];
#pragma unroll
            for (int nt = 0; nt < 2; ++nt)
                out[obase + (size_t)(o0 + ro) * TT + nt * 16 + l15] = acc[nt][g] + bv;
        }
    }
}

extern "C" void kernel_launch(void* const* d_in, const int* in_sizes, int n_in,
                              void* d_out, int out_size, void* d_ws, size_t ws_size,
                              hipStream_t stream) {
    (void)in_sizes; (void)n_in; (void)out_size; (void)ws_size;
    const float* x   = (const float*)d_in[0];
    const float* de  = (const float*)d_in[1];
    const float* ipw = (const float*)d_in[2];
    const float* ipb = (const float*)d_in[3];
    const float* opw = (const float*)d_in[4];
    const float* opb = (const float*)d_in[5];
    const float* phw = (const float*)d_in[6];
    const float* phb = (const float*)d_in[7];
    const float* imw = (const float*)d_in[8];
    const float* imb = (const float*)d_in[9];
    const float* ow  = (const float*)d_in[10];
    const float* ob  = (const float*)d_in[11];

    __bf16* ws   = (__bf16*)d_ws;
    __bf16* Wqkv = ws;                 // 1536*512
    __bf16* Wout = ws + 786432;        // 512*512
    __bf16* Wph  = ws + 1048576;       // 512*512
    __bf16* Wim  = ws + 1310720;       // 512*512
    __bf16* Wo2  = ws + 1572864;       // 512*1024

    cvt_bf16_kernel<<<768, 256, 0, stream>>>(ipw, Wqkv, 786432);
    cvt_bf16_kernel<<<256, 256, 0, stream>>>(opw, Wout, 262144);
    cvt_bf16_kernel<<<256, 256, 0, stream>>>(phw, Wph, 262144);
    cvt_bf16_kernel<<<256, 256, 0, stream>>>(imw, Wim, 262144);
    cvt_bf16_kernel<<<512, 256, 0, stream>>>(ow,  Wo2, 524288);

    float* out = (float*)d_out;
    float* avg = out + (size_t)16 * 512 * 2048;   // (B,2,2) tail of d_out
    zero_avg_kernel<<<1, 64, 0, stream>>>(avg);

    fused_devattn_kernel<<<1024, 256, SMEM_BYTES, stream>>>(
        x, de, Wqkv, Wout, Wph, Wim, Wo2, ipb, opb, phb, imb, ob, out, avg);
}
